// Qwen3OmniCodePredictorAttention_25924422599233
// MI455X (gfx1250) — compile-verified, matrix operand order checked by reading, not run on hardware
//
#include <hip/hip_runtime.h>

// ---------------------------------------------------------------------------
// Qwen3-style GQA attention block for gfx1250 (MI455X), f16 WMMA + f32 acc.
// All matrix math via v_wmma_f32_16x16x32_f16; all fragment feeds are 128-bit.
// ---------------------------------------------------------------------------

typedef __attribute__((ext_vector_type(16))) _Float16 v16h;
typedef __attribute__((ext_vector_type(8)))  _Float16 v8h;
typedef __attribute__((ext_vector_type(8)))  float    v8f;

constexpr int B_   = 2;
constexpr int S_   = 2048;
constexpr int HID_ = 1024;
constexpr int H_   = 16;
constexpr int KV_  = 8;
constexpr int D_   = 64;
constexpr int M_   = B_ * S_;           // 4096 tokens
constexpr float SCALE_ = 0.125f;        // D^-0.5
constexpr float LOG2E_ = 1.4426950408889634f;

__device__ __forceinline__ v8f wmma_f16(v16h a, v16h b, v8f c) {
  // D = A(16x32 f16) * B(32x16 f16) + C(16x16 f32)
  return __builtin_amdgcn_wmma_f32_16x16x32_f16(
      /*neg_a=*/false, a, /*neg_b=*/false, b,
      /*c_mod=*/(short)0, c, /*reuse_a=*/false, /*reuse_b=*/false);
}

// ---------------------------------------------------------------------------
// Weight f32 -> f16 pre-conversion (runs once per call, tiny).
// ---------------------------------------------------------------------------
__global__ void wcvt_kernel(const float* __restrict__ in, _Float16* __restrict__ out, int n) {
  const int i = blockIdx.x * blockDim.x + threadIdx.x;
  if (i < n) out[i] = (_Float16)in[i];
}

// ---------------------------------------------------------------------------
// GEMM: C[M,N] = A[M,K] * W[N,K]^T.  A f32 or f16 -> f16 fragments, W f16,
// f32 WMMA accumulation.  Block = 128 threads = 4 waves; each wave computes a
// 16x64 strip (A-fragment reused 4x).  grid = (M/16, N/256).
// EPI = 0: plain f32 store to C[M,N]                  (output projection)
// EPI = 1: per-head RMSNorm + RoPE, f16 [b,head,s,d]  (Q and K; strip == head)
// EPI = 2: f16 transposed store [b,head,d,s]          (V, feeds P*V B-frags)
// ---------------------------------------------------------------------------
template<bool A_F16, int EPI>
__global__ void gemm_wmma_kernel(const void* __restrict__ Ain,
                                 const _Float16* __restrict__ W,
                                 void* __restrict__ Cout,
                                 const float* __restrict__ cosb,
                                 const float* __restrict__ sinb,
                                 const float* __restrict__ nw,
                                 int M, int N, int K, int NH) {
  const int lane = threadIdx.x & 31;
  const int wv   = threadIdx.x >> 5;       // 0..3
  const int r    = lane & 15;
  const int hi   = lane >> 4;              // half-wave select
  const int m0   = blockIdx.x * 16;
  const int n0   = (blockIdx.y * 4 + wv) * 64;
  const int e0   = hi * 8;                 // K sub-offset per ISA A/B layout

  v8f acc[4];
  #pragma unroll
  for (int nt = 0; nt < 4; ++nt) acc[nt] = (v8f){};

  for (int kb = 0; kb < K; kb += 32) {
    // ---- A fragment: lane = row m0+r, elements K = {e0..e0+7, e0+16..e0+23}
    v16h a;
    if (A_F16) {
      const _Float16* arow = (const _Float16*)Ain + (size_t)(m0 + r) * K + kb + e0;
      v8h lo = *(const v8h*)(arow);
      v8h hv = *(const v8h*)(arow + 16);
      #pragma unroll
      for (int e = 0; e < 8; ++e) { a[e] = lo[e]; a[8 + e] = hv[e]; }
      if (kb + 64 <= K) __builtin_prefetch(arow + 64, 0, 0);   // global_prefetch_b8
    } else {
      const float* arow = (const float*)Ain + (size_t)(m0 + r) * K + kb + e0;
      v8f lo = *(const v8f*)(arow);
      v8f hv = *(const v8f*)(arow + 16);
      #pragma unroll
      for (int e = 0; e < 8; ++e) { a[e] = (_Float16)lo[e]; a[8 + e] = (_Float16)hv[e]; }
      if (kb + 64 <= K) __builtin_prefetch(arow + 64, 0, 0);
    }

    // ---- 4 B fragments: lane = column n, elements along K (W stored [N,K] f16)
    #pragma unroll
    for (int nt = 0; nt < 4; ++nt) {
      const _Float16* wrow = W + (size_t)(n0 + nt * 16 + r) * K + kb + e0;
      v8h lo = *(const v8h*)(wrow);
      v8h hv = *(const v8h*)(wrow + 16);
      v16h bf;
      #pragma unroll
      for (int e = 0; e < 8; ++e) { bf[e] = lo[e]; bf[8 + e] = hv[e]; }
      acc[nt] = wmma_f16(a, bf, acc[nt]);
    }
  }

  // ---- epilogues.  C layout: VGPR j -> row j + 8*hi, lane%16 -> column.
  if (EPI == 0) {
    float* C = (float*)Cout;
    #pragma unroll
    for (int nt = 0; nt < 4; ++nt)
      #pragma unroll
      for (int j = 0; j < 8; ++j)
        C[(size_t)(m0 + j + 8 * hi) * N + n0 + nt * 16 + r] = acc[nt][j];
  } else if (EPI == 1) {
    // Wave strip == one head (64 aligned columns).  Per row: RMS over the 64
    // d-values (4 acc regs x 16 lanes of this half-wave), then RoPE where the
    // (d, d+32) pair is acc[nt] <-> acc[nt+2] in the SAME lane.
    const int head = n0 >> 6;
    _Float16* Of = (_Float16*)Cout;
    #pragma unroll
    for (int j = 0; j < 8; ++j) {
      const int tok = m0 + j + 8 * hi;
      float ss = acc[0][j] * acc[0][j] + acc[1][j] * acc[1][j] +
                 acc[2][j] * acc[2][j] + acc[3][j] * acc[3][j];
      ss += __shfl_xor(ss, 1, 32);
      ss += __shfl_xor(ss, 2, 32);
      ss += __shfl_xor(ss, 4, 32);
      ss += __shfl_xor(ss, 8, 32);
      const float inv = rsqrtf(ss * (1.0f / D_) + 1e-6f);
      float vq[4], cc[4], sn[4];
      const float* cr = cosb + (size_t)tok * D_;
      const float* sr = sinb + (size_t)tok * D_;
      #pragma unroll
      for (int nt = 0; nt < 4; ++nt) {
        vq[nt] = acc[nt][j] * inv * nw[nt * 16 + r];
        cc[nt] = cr[nt * 16 + r];
        sn[nt] = sr[nt * 16 + r];
      }
      const float o0 = vq[0] * cc[0] - vq[2] * sn[0];   // d <  32: -x2*sin
      const float o1 = vq[1] * cc[1] - vq[3] * sn[1];
      const float o2 = vq[2] * cc[2] + vq[0] * sn[2];   // d >= 32: +x1*sin
      const float o3 = vq[3] * cc[3] + vq[1] * sn[3];
      const int b = tok / S_, si = tok % S_;
      _Float16* out = Of + (((size_t)b * NH + head) * S_ + si) * D_;
      out[r]      = (_Float16)o0;
      out[16 + r] = (_Float16)o1;
      out[32 + r] = (_Float16)o2;
      out[48 + r] = (_Float16)o3;
    }
  } else {   // EPI == 2: V, f16 transposed [b, head, d, s]
    const int head = n0 >> 6;
    _Float16* Of = (_Float16*)Cout;
    #pragma unroll
    for (int j = 0; j < 8; ++j) {
      const int tok = m0 + j + 8 * hi;
      const int b = tok / S_, si = tok % S_;
      #pragma unroll
      for (int nt = 0; nt < 4; ++nt) {
        const int d = nt * 16 + r;
        Of[(((size_t)b * NH + head) * D_ + d) * S_ + si] = (_Float16)acc[nt][j];
      }
    }
  }
}

// ---------------------------------------------------------------------------
// Causal flash attention (GQA group=2).  One wave per 16 query rows.
// Scores via 16x16x32 f16 WMMA (Q pre-scaled); online softmax in C-layout;
// P staged through LDS (C-layout -> A-layout); P*V via WMMA, K=32 over t.
// K is [b,kv,s,d]; V is TRANSPOSED [b,kv,d,s] so every fragment feed is b128.
// ---------------------------------------------------------------------------
__global__ void flash_attn_kernel(const _Float16* __restrict__ Q,
                                  const _Float16* __restrict__ Kk,
                                  const _Float16* __restrict__ Vt,
                                  _Float16* __restrict__ O) {
  __shared__ __align__(16) _Float16 ptile[8][16 * 32];   // per-wave 16x32 P tile

  const int lane = threadIdx.x & 31;
  const int wv   = threadIdx.x >> 5;
  const int r    = lane & 15, hi = lane >> 4;
  const int gw   = blockIdx.x * 8 + wv;
  const int qb   = gw & (S_ / 16 - 1);       // 128 q-blocks per (b,h)
  const int bh   = gw >> 7;
  const int h    = bh % H_, b = bh / H_;
  const int kvh  = h / (H_ / KV_);
  const int q0   = qb * 16;
  const int e0   = hi * 8;

  // ---- Q fragments (d-K-steps 0..31 and 32..63), SCALE folded in
  v16h aq0, aq1;
  {
    const _Float16* qrow = Q + (((size_t)b * H_ + h) * S_ + q0 + r) * D_;
    const _Float16 sc = (_Float16)SCALE_;
    v8h q0lo = *(const v8h*)(qrow + e0);
    v8h q0hv = *(const v8h*)(qrow + e0 + 16);
    v8h q1lo = *(const v8h*)(qrow + 32 + e0);
    v8h q1hv = *(const v8h*)(qrow + 32 + e0 + 16);
    #pragma unroll
    for (int e = 0; e < 8; ++e) {
      aq0[e] = q0lo[e] * sc;  aq0[8 + e] = q0hv[e] * sc;
      aq1[e] = q1lo[e] * sc;  aq1[8 + e] = q1hv[e] * sc;
    }
  }

  float rm[8], rs[8];
  v8f acc[4];
  #pragma unroll
  for (int j = 0; j < 8; ++j) { rm[j] = -__builtin_inff(); rs[j] = 0.0f; }
  #pragma unroll
  for (int dt = 0; dt < 4; ++dt) acc[dt] = (v8f){};

  const _Float16* kbase = Kk + ((size_t)b * KV_ + kvh) * (size_t)S_ * D_;
  const _Float16* vbase = Vt + ((size_t)b * KV_ + kvh) * (size_t)D_ * S_;
  _Float16* myp = ptile[wv];

  const int tmax = q0 + 15;
  for (int tb = 0; tb <= tmax; tb += 32) {        // wave-uniform loop
    // ---- scores for two 16-wide t-subtiles
    v8f s0 = (v8f){}, s1 = (v8f){};
    {
      const _Float16* krow = kbase + (size_t)(tb + r) * D_;
      v8h l0 = *(const v8h*)(krow + e0);
      v8h h0 = *(const v8h*)(krow + e0 + 16);
      v8h l1 = *(const v8h*)(krow + 32 + e0);
      v8h h1 = *(const v8h*)(krow + 32 + e0 + 16);
      v16h k0, k1;
      #pragma unroll
      for (int e = 0; e < 8; ++e) {
        k0[e] = l0[e]; k0[8 + e] = h0[e];
        k1[e] = l1[e]; k1[8 + e] = h1[e];
      }
      s0 = wmma_f16(aq0, k0, s0);
      s0 = wmma_f16(aq1, k1, s0);
    }
    if (tb + 16 <= tmax) {                        // wave-uniform branch
      const _Float16* krow = kbase + (size_t)(tb + 16 + r) * D_;
      v8h l0 = *(const v8h*)(krow + e0);
      v8h h0 = *(const v8h*)(krow + e0 + 16);
      v8h l1 = *(const v8h*)(krow + 32 + e0);
      v8h h1 = *(const v8h*)(krow + 32 + e0 + 16);
      v16h k0, k1;
      #pragma unroll
      for (int e = 0; e < 8; ++e) {
        k0[e] = l0[e]; k0[8 + e] = h0[e];
        k1[e] = l1[e]; k1[8 + e] = h1[e];
      }
      s1 = wmma_f16(aq0, k0, s1);
      s1 = wmma_f16(aq1, k1, s1);
    }

    // ---- online softmax over the 32-key block (C-layout rows)
    #pragma unroll
    for (int j = 0; j < 8; ++j) {
      const int row = q0 + j + 8 * hi;
      float a0 = (tb + r      <= row) ? s0[j] : -__builtin_inff();
      float a1 = (tb + 16 + r <= row) ? s1[j] : -__builtin_inff();
      float mx = fmaxf(a0, a1);
      mx = fmaxf(mx, __shfl_xor(mx, 1, 32));
      mx = fmaxf(mx, __shfl_xor(mx, 2, 32));
      mx = fmaxf(mx, __shfl_xor(mx, 4, 32));
      mx = fmaxf(mx, __shfl_xor(mx, 8, 32));
      const float mnew = fmaxf(rm[j], mx);
      const float corr = exp2f((rm[j] - mnew) * LOG2E_);
      const float p0 = exp2f((a0 - mnew) * LOG2E_);
      const float p1 = exp2f((a1 - mnew) * LOG2E_);
      float psum = p0 + p1;
      psum += __shfl_xor(psum, 1, 32);
      psum += __shfl_xor(psum, 2, 32);
      psum += __shfl_xor(psum, 4, 32);
      psum += __shfl_xor(psum, 8, 32);
      rs[j] = rs[j] * corr + psum;
      rm[j] = mnew;
      acc[0][j] *= corr; acc[1][j] *= corr; acc[2][j] *= corr; acc[3][j] *= corr;
      myp[(j + 8 * hi) * 32 + r]      = (_Float16)p0;
      myp[(j + 8 * hi) * 32 + 16 + r] = (_Float16)p1;
    }

    // ---- P A-fragment from LDS (row = lane%16, K = t-within-block)
    v16h pA;
    {
      const _Float16* pl = myp + r * 32;
      v8h lo = *(const v8h*)(pl + e0);
      v8h hv = *(const v8h*)(pl + e0 + 16);
      #pragma unroll
      for (int e = 0; e < 8; ++e) { pA[e] = lo[e]; pA[8 + e] = hv[e]; }
    }

    // ---- P * V (4 d-tiles of 16); V is [d, t] so per-lane reads are b128
    #pragma unroll
    for (int dt = 0; dt < 4; ++dt) {
      const _Float16* vrow = vbase + (size_t)(dt * 16 + r) * S_ + tb;
      v8h lo = *(const v8h*)(vrow + e0);
      v8h hv = *(const v8h*)(vrow + e0 + 16);
      v16h bv;
      #pragma unroll
      for (int e = 0; e < 8; ++e) { bv[e] = lo[e]; bv[8 + e] = hv[e]; }
      acc[dt] = wmma_f16(pA, bv, acc[dt]);
    }
  }

  // ---- epilogue: divide by softmax denom, write f16 [b, s, h, d]
  #pragma unroll
  for (int j = 0; j < 8; ++j) {
    const int row = q0 + j + 8 * hi;
    const float invs = 1.0f / rs[j];
    _Float16* orow = O + (((size_t)b * S_ + row) * H_ + h) * D_;
    orow[r]      = (_Float16)(acc[0][j] * invs);
    orow[16 + r] = (_Float16)(acc[1][j] * invs);
    orow[32 + r] = (_Float16)(acc[2][j] * invs);
    orow[48 + r] = (_Float16)(acc[3][j] * invs);
  }
}

// ---------------------------------------------------------------------------
extern "C" void kernel_launch(void* const* d_in, const int* in_sizes, int n_in,
                              void* d_out, int out_size, void* d_ws, size_t ws_size,
                              hipStream_t stream) {
  (void)in_sizes; (void)n_in; (void)out_size; (void)ws_size;
  const float* hs   = (const float*)d_in[0];   // [B,S,HID]
  const float* cosb = (const float*)d_in[1];   // [B,S,D]
  const float* sinb = (const float*)d_in[2];   // [B,S,D]
  const float* wq   = (const float*)d_in[3];   // [H*D, HID]
  const float* wk   = (const float*)d_in[4];   // [KV*D, HID]
  const float* wv   = (const float*)d_in[5];   // [KV*D, HID]
  const float* wo   = (const float*)d_in[6];   // [HID, H*D]
  const float* qn   = (const float*)d_in[7];   // [D]
  const float* kn   = (const float*)d_in[8];   // [D]

  // Workspace layout (~30 MB, fully rewritten each call):
  char* ws = (char*)d_ws;
  _Float16* Wq16 = (_Float16*)ws;  ws += (size_t)H_  * D_ * HID_ * 2;  // 2 MB
  _Float16* Wk16 = (_Float16*)ws;  ws += (size_t)KV_ * D_ * HID_ * 2;  // 1 MB
  _Float16* Wv16 = (_Float16*)ws;  ws += (size_t)KV_ * D_ * HID_ * 2;  // 1 MB
  _Float16* Wo16 = (_Float16*)ws;  ws += (size_t)HID_ * H_ * D_ * 2;   // 2 MB
  _Float16* Qh   = (_Float16*)ws;  ws += (size_t)M_ * H_  * D_ * 2;    // 8 MB
  _Float16* Kh   = (_Float16*)ws;  ws += (size_t)M_ * KV_ * D_ * 2;    // 4 MB
  _Float16* Vth  = (_Float16*)ws;  ws += (size_t)M_ * KV_ * D_ * 2;    // 4 MB
  _Float16* Ah   = (_Float16*)ws;                                      // 8 MB

  // Weight pre-conversion to f16 (one-time per call, negligible cost)
  wcvt_kernel<<<(H_  * D_ * HID_) / 256, 256, 0, stream>>>(wq, Wq16, H_  * D_ * HID_);
  wcvt_kernel<<<(KV_ * D_ * HID_) / 256, 256, 0, stream>>>(wk, Wk16, KV_ * D_ * HID_);
  wcvt_kernel<<<(KV_ * D_ * HID_) / 256, 256, 0, stream>>>(wv, Wv16, KV_ * D_ * HID_);
  wcvt_kernel<<<(HID_ * H_ * D_) / 256, 256, 0, stream>>>(wo, Wo16, HID_ * H_ * D_);

  const dim3 blk(128);
  // QKV projections with fused epilogues (RMSNorm+RoPE for Q/K, transpose for V)
  gemm_wmma_kernel<false, 1><<<dim3(M_ / 16, (H_  * D_) / 256), blk, 0, stream>>>(
      hs, Wq16, Qh, cosb, sinb, qn, M_, H_  * D_, HID_, H_);
  gemm_wmma_kernel<false, 1><<<dim3(M_ / 16, (KV_ * D_) / 256), blk, 0, stream>>>(
      hs, Wk16, Kh, cosb, sinb, kn, M_, KV_ * D_, HID_, KV_);
  gemm_wmma_kernel<false, 2><<<dim3(M_ / 16, (KV_ * D_) / 256), blk, 0, stream>>>(
      hs, Wv16, Vth, nullptr, nullptr, nullptr, M_, KV_ * D_, HID_, KV_);

  // Causal flash attention (GQA)
  flash_attn_kernel<<<(B_ * H_ * (S_ / 16)) / 8, 256, 0, stream>>>(Qh, Kh, Vth, Ah);

  // Output projection, writes f32 result directly to d_out
  gemm_wmma_kernel<true, 0><<<dim3(M_ / 16, HID_ / 256), blk, 0, stream>>>(
      Ah, Wo16, (float*)d_out, nullptr, nullptr, nullptr, M_, HID_, HID_, H_);
}
